// TemporalBlock_21208548508048
// MI455X (gfx1250) — matrix operand order repaired and checked by reading, not compiled, not run on hardware
//
#include <hip/hip_runtime.h>

// CDNA5 / gfx1250, wave32. Fused GEMM(f16 WMMA) + LIF(tau=2, vth=1, hard reset)
// + multiplicative spike gate. Binary intermediates stored as u8 (ping-pong in
// d_ws). Weights pre-converted to f16 once; each block keeps its whole B panel
// resident in LDS across all timesteps; A tiles double-buffered in LDS.

typedef __attribute__((ext_vector_type(16))) _Float16 v16h;
typedef __attribute__((ext_vector_type(8)))  _Float16 v8h;
typedef __attribute__((ext_vector_type(8)))  float    v8f;

#define TSTEPS   4
#define BN_ROWS  32768   // B*N = 32*1024
#define BM       128     // block tile rows (M)
#define BNCOL    128     // block tile cols (N = Dout tile)
#define BK       32      // K per WMMA step
#define LDSTR    40      // 32 + 8 halfs pad (80B rows, 16B aligned)

// ---- stage 16 elements -> f16, converting from the layer's storage type ----
__device__ __forceinline__ void load16_cvt(const float* __restrict__ p,
                                           _Float16* __restrict__ o) {
  const float4* q = (const float4*)p;
#pragma unroll
  for (int i = 0; i < 4; ++i) {
    float4 f = q[i];
    o[i*4+0] = (_Float16)f.x; o[i*4+1] = (_Float16)f.y;
    o[i*4+2] = (_Float16)f.z; o[i*4+3] = (_Float16)f.w;
  }
}
__device__ __forceinline__ void load16_cvt(const unsigned char* __restrict__ p,
                                           _Float16* __restrict__ o) {
  uint4 u = *(const uint4*)p;
  unsigned int w[4] = {u.x, u.y, u.z, u.w};
#pragma unroll
  for (int i = 0; i < 16; ++i)
    o[i] = (_Float16)(float)((w[i >> 2] >> ((i & 3) * 8)) & 0xffu);
}

__device__ __forceinline__ v16h cat8(v8h lo, v8h hi) {
  v16h r;
#pragma unroll
  for (int j = 0; j < 8; ++j) { r[j] = lo[j]; r[j + 8] = hi[j]; }
  return r;
}

// one-shot fp32 -> f16 weight conversion
__global__ __launch_bounds__(256)
void cvt_w_f16(const float* __restrict__ w, _Float16* __restrict__ o, int n) {
  int i = blockIdx.x * 256 + threadIdx.x;
  if (i < n) o[i] = (_Float16)w[i];
}

// One layer: out[t,m,n] = LIF over t of (act[t] @ wgt^T), optionally gated:
// out = gate * (1 - spike).
template <int DIN, bool GATE, typename TIn, typename TOut>
__global__ __launch_bounds__(256)
void snn_lif_wmma(const TIn* __restrict__ act,            // [T, BN_ROWS, DIN]
                  const _Float16* __restrict__ wgt,       // [Dout, DIN] f16
                  TOut* __restrict__ out,                 // [T, BN_ROWS, Dout]
                  const unsigned char* __restrict__ gate, // == act buffer if GATE
                  int Dout) {
  constexpr int BSTR = DIN + 8;                 // padded B-panel row stride
  __shared__ _Float16 sA[2][BM][LDSTR];         // double-buffered A tile
  __shared__ _Float16 sB[BNCOL][BSTR];          // resident B panel (all K)

  const int tid   = threadIdx.x;
  const int lane  = tid & 31;
  const int wave  = tid >> 5;
  const int waveM = (wave & 3) * 32;   // 4 waves along M (32 rows each)
  const int waveN = (wave >> 2) * 64;  // 2 waves along N (64 cols each)
  const int lhalf = lane >> 4;         // half-wave select (ISA frag layouts)
  const int lmod  = lane & 15;

  const int blockM = blockIdx.x * BM;
  const int blockN = blockIdx.y * BNCOL;

  // staging: 256 threads cover 128 rows x (2 halves of 16 elements)
  const int ldRow  = tid >> 1;
  const int ldKOff = (tid & 1) * 16;

  // ---- stage the whole f16 B panel for this column tile, once ----
  {
    constexpr int CHUNKS = (BNCOL * DIN) / 16;  // 16-half chunks
    constexpr int PER_TH = CHUNKS / 256;        // DIN/32
#pragma unroll
    for (int i = 0; i < PER_TH; ++i) {
      int c   = tid + i * 256;
      int row = c / (DIN / 16);
      int ko  = (c % (DIN / 16)) * 16;
      const v8h* src = (const v8h*)(wgt + (size_t)(blockN + row) * DIN + ko);
      *(v8h*)&sB[row][ko]     = src[0];
      *(v8h*)&sB[row][ko + 8] = src[1];
    }
  }

  // persistent LIF membrane state (f32), one per owned output element
  v8f vst[2][4];
#pragma unroll
  for (int mi = 0; mi < 2; ++mi)
#pragma unroll
    for (int ni = 0; ni < 4; ++ni)
#pragma unroll
      for (int r = 0; r < 8; ++r) vst[mi][ni][r] = 0.0f;

  for (int t = 0; t < TSTEPS; ++t) {
    const TIn* aRow =
        act + ((size_t)t * BN_ROWS + blockM + ldRow) * (size_t)DIN + ldKOff;

    v8f acc[2][4];
#pragma unroll
    for (int mi = 0; mi < 2; ++mi)
#pragma unroll
      for (int ni = 0; ni < 4; ++ni)
#pragma unroll
        for (int r = 0; r < 8; ++r) acc[mi][ni][r] = 0.0f;

    auto stageA = [&](int buf, int k0) {
      __align__(16) _Float16 tmp[16];
      load16_cvt(aRow + k0, tmp);
      *(v8h*)&sA[buf][ldRow][ldKOff]     = *(const v8h*)&tmp[0];
      *(v8h*)&sA[buf][ldRow][ldKOff + 8] = *(const v8h*)&tmp[8];
    };

    __syncthreads();              // prev-t LDS reads (or B staging) complete
    stageA(0, 0);
    int buf = 0;

    for (int k0 = 0; k0 < DIN; k0 += BK) {
      __syncthreads();            // sA[buf] visible block-wide
      if (k0 + BK < DIN) {
        __builtin_prefetch(aRow + k0 + 2 * BK, 0, 0);   // global_prefetch_b8
        stageA(buf ^ 1, k0 + BK); // overlap next tile with current WMMAs
      }

      // A fragment (16-bit 16x32): row = lmod, K = [8*lhalf,+8) U [+16,+24)
      v16h afr[2];
#pragma unroll
      for (int mi = 0; mi < 2; ++mi) {
        const _Float16* ar = &sA[buf][waveM + mi * 16 + lmod][0];
        afr[mi] = cat8(*(const v8h*)(ar + lhalf * 8),
                       *(const v8h*)(ar + lhalf * 8 + 16));
      }
      // B fragment (32x16): col = lmod, K = k0 + [8*lhalf,+8) U [+16,+24)
      v16h bfr[4];
#pragma unroll
      for (int ni = 0; ni < 4; ++ni) {
        const _Float16* br = &sB[waveN + ni * 16 + lmod][k0];
        bfr[ni] = cat8(*(const v8h*)(br + lhalf * 8),
                       *(const v8h*)(br + lhalf * 8 + 16));
      }
#pragma unroll
      for (int mi = 0; mi < 2; ++mi)
#pragma unroll
        for (int ni = 0; ni < 4; ++ni)
          acc[mi][ni] = __builtin_amdgcn_wmma_f32_16x16x32_f16(
              false, afr[mi], false, bfr[ni], (short)0, acc[mi][ni],
              false, false);
      buf ^= 1;
    }

    // LIF epilogue: v = (v + y)/2 ; s = H(v - 1) ; v *= (1 - s); write spikes.
    // D layout: VGPR r -> M = r + 8*lhalf, N = lmod.
#pragma unroll
    for (int mi = 0; mi < 2; ++mi) {
#pragma unroll
      for (int ni = 0; ni < 4; ++ni) {
        const int n = blockN + waveN + ni * 16 + lmod;
#pragma unroll
        for (int r = 0; r < 8; ++r) {
          const int m = blockM + waveM + mi * 16 + lhalf * 8 + r;
          float v = 0.5f * (vst[mi][ni][r] + acc[mi][ni][r]);
          int   s = (v >= 1.0f) ? 1 : 0;
          vst[mi][ni][r] = s ? 0.0f : v;
          size_t idx = ((size_t)t * BN_ROWS + m) * (size_t)Dout + n;
          if constexpr (GATE) {
            out[idx] = (TOut)(gate[idx] & (unsigned char)(1 - s));
          } else {
            out[idx] = (TOut)s;
          }
        }
      }
    }
  }
}

extern "C" void kernel_launch(void* const* d_in, const int* in_sizes, int n_in,
                              void* d_out, int out_size, void* d_ws,
                              size_t ws_size, hipStream_t stream) {
  (void)in_sizes; (void)n_in; (void)out_size; (void)ws_size;

  const float* x     = (const float*)d_in[0];  // [4,32,1024,256] f32
  const float* w_in  = (const float*)d_in[1];  // [512,256]
  const float* w_h   = (const float*)d_in[2];  // [2,512,512]
  const float* w_out = (const float*)d_in[3];  // [256,512]

  // ws layout: two 64 MiB u8 activation buffers + 1.5 MiB f16 weights
  unsigned char* hA = (unsigned char*)d_ws;
  unsigned char* hB = hA + 67108864ull;
  _Float16* wf_in  = (_Float16*)(hB + 67108864ull);
  _Float16* wf_h0  = wf_in + 131072;     // 512*256
  _Float16* wf_h1  = wf_h0 + 262144;     // 512*512
  _Float16* wf_out = wf_h1 + 262144;     // 512*512 -> then 256*512

  dim3 blk(256);
  cvt_w_f16<<<dim3(131072 / 256), blk, 0, stream>>>(w_in, wf_in, 131072);
  cvt_w_f16<<<dim3(524288 / 256), blk, 0, stream>>>(w_h, wf_h0, 524288);
  cvt_w_f16<<<dim3(131072 / 256), blk, 0, stream>>>(w_out, wf_out, 131072);

  dim3 gM4(BN_ROWS / BM, 512 / BNCOL);  // Dout=512
  dim3 gM2(BN_ROWS / BM, 256 / BNCOL);  // Dout=256

  // h0 = LIF(x @ w_in^T)
  snn_lif_wmma<256, false, float, unsigned char>
      <<<gM4, blk, 0, stream>>>(x, wf_in, hA, nullptr, 512);
  // h1 = h0 * (1 - LIF(h0 @ w_h[0]^T))
  snn_lif_wmma<512, true, unsigned char, unsigned char>
      <<<gM4, blk, 0, stream>>>(hA, wf_h0, hB, hA, 512);
  // h2 = h1 * (1 - LIF(h1 @ w_h[1]^T))
  snn_lif_wmma<512, true, unsigned char, unsigned char>
      <<<gM4, blk, 0, stream>>>(hB, wf_h1, hA, hB, 512);
  // out = LIF(h2 @ w_out^T)  (fp32)
  snn_lif_wmma<512, false, unsigned char, float>
      <<<gM2, blk, 0, stream>>>(hA, wf_out, (float*)d_out, nullptr, 256);
}
